// SigLIPLoss_74208444940927
// MI455X (gfx1250) — matrix operand order repaired and checked by reading, not compiled, not run on hardware
//
#include <hip/hip_runtime.h>
#include <hip/hip_bf16.h>
#include <math.h>

typedef _Float16 v4h  __attribute__((ext_vector_type(4)));
typedef _Float16 v8h  __attribute__((ext_vector_type(8)));
typedef _Float16 v16h __attribute__((ext_vector_type(16)));
typedef float    v8f  __attribute__((ext_vector_type(8)));

#define DIM 512          // embedding dim (fixed by reference)
#define KSTEPS (DIM / 32)

// ---------------------------------------------------------------------------
// Kernel 1: L2-normalize rows of both embedding matrices, convert to f16.
// One wave (32 lanes) per row; 512 floats -> 16 per lane.
// ---------------------------------------------------------------------------
__global__ __launch_bounds__(256) void siglip_normalize_kernel(
    const float* __restrict__ img, const float* __restrict__ prof,
    _Float16* __restrict__ imgH, _Float16* __restrict__ profH, int nrows)
{
  int wave = (blockIdx.x * blockDim.x + threadIdx.x) >> 5;
  int lane = threadIdx.x & 31;
  const float* src;
  _Float16*    dst;
  int row;
  if (wave < nrows) { src = img;  dst = imgH;  row = wave; }
  else              { src = prof; dst = profH; row = wave - nrows; }

  const float4* s4 = (const float4*)(src + (size_t)row * DIM);
  float4 v[4];
  float ss = 0.f;
#pragma unroll
  for (int c = 0; c < 4; ++c) {
    v[c] = s4[c * 32 + lane];
    ss += v[c].x * v[c].x + v[c].y * v[c].y + v[c].z * v[c].z + v[c].w * v[c].w;
  }
#pragma unroll
  for (int off = 16; off > 0; off >>= 1)
    ss += __shfl_xor(ss, off, 32);

  float inv = 1.0f / fmaxf(sqrtf(ss), 1e-12f);

  v4h* d4 = (v4h*)(dst + (size_t)row * DIM);
#pragma unroll
  for (int c = 0; c < 4; ++c) {
    v4h h;
    h.x = (_Float16)(v[c].x * inv);
    h.y = (_Float16)(v[c].y * inv);
    h.z = (_Float16)(v[c].z * inv);
    h.w = (_Float16)(v[c].w * inv);
    d4[c * 32 + lane] = h;
  }
}

// ---------------------------------------------------------------------------
// Kernel 2: WMMA GEMM over 64x64 logits tiles + fused SigLIP epilogue.
// Each wave: 4x4 grid of 16x16 f32 accumulators, K=512 via 16 steps of
// v_wmma_f32_16x16x32_f16. One deterministic partial sum per tile.
// ---------------------------------------------------------------------------
__global__ __launch_bounds__(256) void siglip_gemm_kernel(
    const _Float16* __restrict__ A,   // normalized image emb, f16, n x 512
    const _Float16* __restrict__ B,   // normalized profile emb, f16, n x 512
    const float* __restrict__ logit_scale_p,
    const float* __restrict__ bias_p,
    float* __restrict__ partials, int tilesPerDim)
{
  const int lane    = threadIdx.x & 31;
  const int waveBlk = threadIdx.x >> 5;
  const int tile    = blockIdx.x * 8 + waveBlk;
  const int tr      = tile / tilesPerDim;
  const int tc      = tile % tilesPerDim;
  const int rowBase = tr * 64;
  const int colBase = tc * 64;

  const int  l15  = lane & 15;
  const bool hsel = (lane >= 16);

  const float scale = expf(logit_scale_p[0]);
  const float bias  = bias_p[0];

  v8f acc[4][4];
#pragma unroll
  for (int i = 0; i < 4; ++i)
#pragma unroll
    for (int j = 0; j < 4; ++j)
      acc[i][j] = (v8f){0.f, 0.f, 0.f, 0.f, 0.f, 0.f, 0.f, 0.f};

  // Per-lane row bases (A: M = l%16 in each 16-row sub-tile; B: N = l%16).
  const _Float16* aBase = A + (size_t)(rowBase + l15) * DIM + (hsel ? 8 : 0);
  const _Float16* bBase = B + (size_t)(colBase + l15) * DIM + (hsel ? 8 : 0);

  for (int kb = 0; kb < KSTEPS; ++kb) {
    const int k0 = kb * 32;
    v16h a[4], b[4];
#pragma unroll
    for (int i = 0; i < 4; ++i) {
      // A 16x32 f16 layout: lanes 0-15 hold K 0-7 (v0-3) & 16-23 (v4-7);
      // lanes 16-31 hold K 8-15 & 24-31.  Two 16B chunks, 32B apart? No: +16 halves.
      const _Float16* p = aBase + (size_t)i * 16 * DIM + k0;
      v8h lo = *(const v8h*)p;
      v8h hi = *(const v8h*)(p + 16);
      a[i] = __builtin_shufflevector(lo, hi, 0, 1, 2, 3, 4, 5, 6, 7,
                                             8, 9, 10, 11, 12, 13, 14, 15);
    }
#pragma unroll
    for (int j = 0; j < 4; ++j) {
      // B 32x16 f16 layout: lanes 0-15 hold K 0-7 & 16-23 of column n;
      // lanes 16-31 hold K 8-15 & 24-31 (same order as A).
      const _Float16* p = bBase + (size_t)j * 16 * DIM + k0;
      v8h lo = *(const v8h*)p;
      v8h hi = *(const v8h*)(p + 16);
      b[j] = __builtin_shufflevector(lo, hi, 0, 1, 2, 3, 4, 5, 6, 7,
                                             8, 9, 10, 11, 12, 13, 14, 15);
    }
#pragma unroll
    for (int i = 0; i < 4; ++i)
#pragma unroll
      for (int j = 0; j < 4; ++j)
        acc[i][j] = __builtin_amdgcn_wmma_f32_16x16x32_f16(
            false, a[i], false, b[j], (short)0, acc[i][j], false, false);
  }

  // Fused epilogue: z = diag ? logit : -logit ; accumulate softplus(-z).
  float local = 0.f;
#pragma unroll
  for (int i = 0; i < 4; ++i)
#pragma unroll
    for (int j = 0; j < 4; ++j)
#pragma unroll
      for (int r = 0; r < 8; ++r) {
        const float s   = acc[i][j][r];
        const int   row = rowBase + 16 * i + r + (hsel ? 8 : 0);
        const int   col = colBase + 16 * j + l15;
        const float lg  = fmaf(s, scale, bias);
        const float z   = (row == col) ? lg : -lg;
        const float x   = -z;
        local += fmaxf(x, 0.f) + log1pf(expf(-fabsf(x)));
      }

#pragma unroll
  for (int off = 16; off > 0; off >>= 1)
    local += __shfl_down(local, off, 32);
  if (lane == 0) partials[tile] = local;
}

// ---------------------------------------------------------------------------
// Kernel 3: deterministic reduction of tile partials -> final scalar loss.
// ---------------------------------------------------------------------------
__global__ __launch_bounds__(256) void siglip_reduce_kernel(
    const float* __restrict__ partials, float* __restrict__ out,
    int nPartials, float invBs)
{
  __shared__ float sm[256];
  float s = 0.f;
  for (int i = threadIdx.x; i < nPartials; i += 256) s += partials[i];
  sm[threadIdx.x] = s;
  __syncthreads();
  for (int off = 128; off > 0; off >>= 1) {
    if ((int)threadIdx.x < off) sm[threadIdx.x] += sm[threadIdx.x + off];
    __syncthreads();
  }
  if (threadIdx.x == 0) out[0] = sm[0] * invBs;
}

// ---------------------------------------------------------------------------
extern "C" void kernel_launch(void* const* d_in, const int* in_sizes, int n_in,
                              void* d_out, int out_size, void* d_ws, size_t ws_size,
                              hipStream_t stream) {
  const float* img   = (const float*)d_in[0];
  const float* prof  = (const float*)d_in[1];
  const float* lscal = (const float*)d_in[2];
  const float* bias  = (const float*)d_in[3];
  // d_in[4] = buckets (==1 per reference)

  const int n = in_sizes[0] / DIM;              // 16384
  const int tilesPerDim = n / 64;               // 256
  const int nTiles = tilesPerDim * tilesPerDim; // 65536

  char* ws = (char*)d_ws;
  _Float16* imgH  = (_Float16*)ws;
  _Float16* profH = (_Float16*)(ws + (size_t)n * DIM * sizeof(_Float16));
  float* partials = (float*)(ws + 2 * (size_t)n * DIM * sizeof(_Float16));

  // 1) normalize both matrices: 2*n rows, 1 wave/row, 8 waves/block
  siglip_normalize_kernel<<<(2 * n) / 8, 256, 0, stream>>>(img, prof, imgH, profH, n);

  // 2) GEMM + fused loss epilogue: 8 wave-tiles per 256-thread block
  siglip_gemm_kernel<<<nTiles / 8, 256, 0, stream>>>(imgH, profH, lscal, bias,
                                                     partials, tilesPerDim);

  // 3) final deterministic reduction; loss = sum / bs (buckets == 1)
  siglip_reduce_kernel<<<1, 256, 0, stream>>>(partials, (float*)d_out, nTiles,
                                              1.0f / (float)n);
}
